// MultiheadFlashDiff_42717744726610
// MI455X (gfx1250) — matrix operand order repaired and checked by reading, not compiled, not run on hardware
//
#include <hip/hip_runtime.h>
#include <hip/hip_bf16.h>
#include <stddef.h>
#include <stdint.h>

// ---------------------------------------------------------------------------
// MultiheadFlashDiff on gfx1250 (MI455X): bf16 WMMA GEMMs (async-LDS staged)
// + WMMA flash attention with TDM (tensor_load_to_lds) K/V tile staging.
// Shapes: B=2, S=2048, D=2048, H=16, hd=64 (qk head dim), dv=128 (v head dim)
// ---------------------------------------------------------------------------

typedef __bf16 bf16;
typedef __bf16 v16bf __attribute__((ext_vector_type(16)));
typedef __bf16 v8bf  __attribute__((ext_vector_type(8)));
typedef float  v8f   __attribute__((ext_vector_type(8)));
typedef unsigned int v4u  __attribute__((ext_vector_type(4)));
typedef int          v4i_ __attribute__((ext_vector_type(4)));
typedef int          v8i_ __attribute__((ext_vector_type(8)));

#define BATCH   2
#define SEQ     2048
#define DMODEL  2048
#define NHEADS  16
#define HDQK    64
#define HDV     128
#define NROWS   (BATCH * SEQ)           // 4096
#define LAMBDA_INIT_F   0.7836057665316245f
#define ONE_MINUS_LI_F  0.21639423346837548f

// LDS row pitches (bytes): multiples of 16 (DS B128 alignment), and chosen so
// 16 consecutive rows hit distinct bank groups (144=36w, 80=20w vs 64 banks).
// These pitches are also exactly what TDM pad_enable produces (see below).
#define KPITCH  144     // K tile rows: 128B data + 16B pad
#define VPITCH  80      // V tile rows:  64B data + 16B pad
#define BPITCH  80      // GEMM B tile rows: 64B data + 16B pad
#define KTILE_B (32 * KPITCH)     // 4608
#define VTILE_B (128 * VPITCH)    // 10240
#define BTILE_B (64 * BPITCH)     // 5120

#if __has_builtin(__builtin_amdgcn_tensor_load_to_lds)
#define HAVE_TDM 1
#else
#define HAVE_TDM 0
#endif

__device__ __forceinline__ v8f wmma_bf16(v16bf a, v16bf b, v8f c) {
  // D = A(16x32 bf16) * B(32x16 bf16) + C(16x16 f32)
  return __builtin_amdgcn_wmma_f32_16x16x32_bf16(
      /*neg_a=*/false, a, /*neg_b=*/false, b,
      /*c_mod=*/(short)0, c, /*reuse_a=*/false, /*reuse_b=*/false);
}

__device__ __forceinline__ v16bf cat8(v8bf lo, v8bf hi) {
  return __builtin_shufflevector(lo, hi, 0,1,2,3,4,5,6,7,8,9,10,11,12,13,14,15);
}

__device__ __forceinline__ void wait_asynccnt0() {
#if __has_builtin(__builtin_amdgcn_s_wait_asynccnt)
  __builtin_amdgcn_s_wait_asynccnt(0);
#else
  asm volatile("s_wait_asynccnt 0" ::: "memory");
#endif
}

__device__ __forceinline__ void wait_tensorcnt0() {
#if __has_builtin(__builtin_amdgcn_s_wait_tensorcnt)
  __builtin_amdgcn_s_wait_tensorcnt(0);
#else
  asm volatile("s_wait_tensorcnt 0" ::: "memory");
#endif
}

// Async copy of 16 bytes per lane: global -> LDS (ASYNCcnt-tracked).
__device__ __forceinline__ void async_b128(uint32_t lds_off, const void* gptr) {
  unsigned long long ga = (unsigned long long)(uintptr_t)gptr;
  uint32_t l = lds_off;
  asm volatile("global_load_async_to_lds_b128 %0, %1, off"
               :: "v"(l), "v"(ga) : "memory");
}

#if HAVE_TDM
// TDM 2D tile load: global (row-major, stride0 elements of 2B) -> LDS with
// hardware row padding. D# built per CDNA5 ISA 8.3/8.4:
//  group0: count=1 | lds_addr | global_addr[56:0] | type=2
//  group1: data_size=2B, pad_enable, pad_interval/pad_amount codes,
//          tensor_dim0/1, tile_dim0/1, tensor_dim0_stride
//  groups 2/3: zeros (2D tensor; tile_dim3/4=0 -> unused)
__device__ __forceinline__ void tdm_load_2d(uint32_t lds_off, const void* gaddr,
                                            uint32_t tile_d0, uint32_t tile_d1,
                                            uint32_t tensor_d0, uint32_t tensor_d1,
                                            uint32_t stride0,
                                            uint32_t pad_interval_code,
                                            uint32_t pad_amount_code) {
  const unsigned long long ga = (unsigned long long)(uintptr_t)gaddr;
  v4u g0;
  g0[0] = 1u;                                           // count=1, user mode
  g0[1] = lds_off;                                      // lds_addr
  g0[2] = (uint32_t)ga;                                 // global_addr[31:0]
  g0[3] = ((uint32_t)(ga >> 32) & 0x01FFFFFFu)          // global_addr[56:32]
          | 0x80000000u;                                // type=2 ("image")
  v8i_ g1;
  g1[0] = (int)((1u << 16)                              // data_size=1 -> 2B
                | (1u << 20)                            // pad_enable
                | (pad_interval_code << 22)
                | (pad_amount_code << 25));
  g1[1] = (int)((tensor_d0 & 0xFFFFu) << 16);           // tensor_dim0[15:0]
  g1[2] = (int)(((tensor_d0 >> 16) & 0xFFFFu)           // tensor_dim0[31:16]
                | ((tensor_d1 & 0xFFFFu) << 16));       // tensor_dim1[15:0]
  g1[3] = (int)(((tensor_d1 >> 16) & 0xFFFFu)           // tensor_dim1[31:16]
                | (tile_d0 << 16));                     // tile_dim0
  g1[4] = (int)tile_d1;                                 // tile_dim1 (tile_dim2=0)
  g1[5] = (int)stride0;                                 // tensor_dim0_stride lo
  g1[6] = 0;                                            // stride0 hi, dim1_stride lo
  g1[7] = 0;
  v4i_ z = {0, 0, 0, 0};
#if __clang_major__ >= 23
  v8i_ z8 = {0, 0, 0, 0, 0, 0, 0, 0};
  __builtin_amdgcn_tensor_load_to_lds(g0, g1, z, z, z8, 0);
#else
  __builtin_amdgcn_tensor_load_to_lds(g0, g1, z, z, 0);
#endif
}
#endif  // HAVE_TDM

// Read one 32x16 B-fragment from an LDS tile with row pitch `pitch` bytes:
// lane<16 -> col=lane, elems 0-7 = K 0-7, 8-15 = K 16-23; lane>=16 -> K 8-15, 24-31.
__device__ __forceinline__ v16bf lds_b_frag(const char* tile, int col_base,
                                            int kbyte0, int pitch, int lane) {
  const char* p = tile + (size_t)(col_base + (lane & 15)) * pitch
                       + kbyte0 + ((lane >> 4) << 4);
  v8bf lo = *(const v8bf*)p;
  v8bf hi = *(const v8bf*)(p + 32);
  return cat8(lo, hi);
}

// A fragment: 16x32 bf16, row-major global source with leading dim lda.
__device__ __forceinline__ v16bf load_a_frag(const bf16* A, int lda,
                                             int row_base, int k0, int lane) {
  int row = row_base + (lane & 15);
  int kb  = k0 + ((lane >> 4) << 3);
  const bf16* p = A + (size_t)row * lda + kb;
  v8bf lo = *(const v8bf*)p;
  v8bf hi = *(const v8bf*)(p + 16);
  return cat8(lo, hi);
}

// ---------------------------------------------------------------------------
// Prep kernels
// ---------------------------------------------------------------------------
__global__ void k_f32_to_bf16(const float* __restrict__ in,
                              bf16* __restrict__ out, int n) {
  int i = blockIdx.x * blockDim.x + threadIdx.x;
  if (i < n) out[i] = (bf16)in[i];
}

// Wt[n][k] = (bf16) W[k][n], 2048x2048, LDS-tiled (coalesced both sides)
__global__ void k_transpose_w(const float* __restrict__ W,
                              bf16* __restrict__ Wt) {
  __shared__ bf16 t[32][33];
  const int kb = blockIdx.x * 32;       // k tile base
  const int nb = blockIdx.y * 32;       // n tile base
  const int tx = threadIdx.x, ty = threadIdx.y;   // 32 x 8
#pragma unroll
  for (int i = 0; i < 32; i += 8)
    t[ty + i][tx] = (bf16)W[(size_t)(kb + ty + i) * DMODEL + nb + tx];
  __syncthreads();
#pragma unroll
  for (int i = 0; i < 32; i += 8)
    Wt[(size_t)(nb + ty + i) * DMODEL + kb + tx] = t[tx][ty + i];
}

// ---------------------------------------------------------------------------
// GEMM mainloop: 4-wave block computes 64(M) x 64(N), K=2048.
// B panel (64 cols x 32 k bf16) async-staged in LDS (pitch 80B), double
// buffered; A streamed per-wave straight from global.
// ---------------------------------------------------------------------------
__device__ __forceinline__ void async_fill_b(const bf16* __restrict__ Bt,
                                             int n0, int k0,
                                             char* bs_buf, int tid) {
  const uint32_t lbase = (uint32_t)(uintptr_t)bs_buf;   // low 32 = LDS offset
#pragma unroll
  for (int half = 0; half < 2; ++half) {
    const int ci = tid + half * 128;            // 256 x 16B chunks, 4 per row
    const int row = ci >> 2, inrow = (ci & 3) * 16;
    const char* g = (const char*)(Bt + (size_t)(n0 + row) * DMODEL + k0) + inrow;
    async_b128(lbase + row * BPITCH + inrow, g);
  }
}

__device__ __forceinline__ void gemm_mainloop(const bf16* __restrict__ A,
                                              const bf16* __restrict__ Bt,
                                              char* Bs,        // [2][BTILE_B]
                                              int m0, int n0,
                                              int tid, int lane, v8f acc[4]) {
  async_fill_b(Bt, n0, 0, Bs, tid);
  int c = 0;
  for (int k0 = 0; k0 < DMODEL; k0 += 32) {
    wait_asynccnt0();
    __syncthreads();                       // buffer c ready for everyone
    if (k0 + 32 < DMODEL)
      async_fill_b(Bt, n0, k0 + 32, Bs + (1 - c) * BTILE_B, tid);
    v16bf a = load_a_frag(A, DMODEL, m0, k0, lane);
    const char* tile = Bs + c * BTILE_B;
#pragma unroll
    for (int j = 0; j < 4; ++j)
      acc[j] = wmma_bf16(a, lds_b_frag(tile, 16 * j, 0, BPITCH, lane), acc[j]);
    __syncthreads();                       // all reads of buffer c done
    c ^= 1;
  }
}

// C(f32, row-major 4096x2048) epilogue
__global__ void __launch_bounds__(128)
k_gemm_f32(const bf16* __restrict__ A, const bf16* __restrict__ Bt,
           float* __restrict__ C) {
  __shared__ char Bs[2 * BTILE_B];
  const int tid  = threadIdx.x;
  const int lane = tid & 31;
  const int wave = tid >> 5;
  const int m0 = blockIdx.x * 64 + wave * 16;
  const int n0 = blockIdx.y * 64;
  v8f acc[4] = {};
  gemm_mainloop(A, Bt, Bs, m0, n0, tid, lane, acc);

  const int rbase = m0 + ((lane >> 4) << 3);
  const int cl = lane & 15;
#pragma unroll
  for (int j = 0; j < 4; ++j)
#pragma unroll
    for (int r = 0; r < 8; ++r)
      C[(size_t)(rbase + r) * DMODEL + n0 + 16 * j + cl] = acc[j][r];
}

// V projection: epilogue writes bf16 V transposed: Vt[b][h][vd][s]
__global__ void __launch_bounds__(128)
k_gemm_vt(const bf16* __restrict__ A, const bf16* __restrict__ Bt,
          bf16* __restrict__ Vt) {
  __shared__ char Bs[2 * BTILE_B];
  const int tid  = threadIdx.x;
  const int lane = tid & 31;
  const int wave = tid >> 5;
  const int m0 = blockIdx.x * 64 + wave * 16;
  const int n0 = blockIdx.y * 64;
  v8f acc[4] = {};
  gemm_mainloop(A, Bt, Bs, m0, n0, tid, lane, acc);

  const int Mb = m0 + ((lane >> 4) << 3);     // 8 consecutive rows per lane
  const int bb = Mb >> 11;                    // batch
  const int ss = Mb & 2047;                   // seq pos
#pragma unroll
  for (int j = 0; j < 4; ++j) {
    const int n = n0 + 16 * j + (lane & 15);  // global v column
    const int h = n >> 7, vd = n & 127;
    v8bf o;
#pragma unroll
    for (int r = 0; r < 8; ++r) o[r] = (bf16)acc[j][r];
    *(v8bf*)(Vt + (((size_t)bb * NHEADS + h) * HDV + vd) * SEQ + ss) = o;
  }
}

// ---------------------------------------------------------------------------
// RoPE (interleaved) fp32 -> bf16, layout (4096 rows x 2048 cols)
// ---------------------------------------------------------------------------
__global__ void k_rope(const float* __restrict__ in, bf16* __restrict__ out) {
  int p = blockIdx.x * blockDim.x + threadIdx.x;   // pair index
  int row = p >> 10;             // b*2048 + s
  int cp  = p & 1023;
  int ch  = cp >> 5;             // channel (0..31)
  int j   = cp & 31;             // pair within head dim
  int col = ch * 64 + 2 * j;
  float t = (float)(row & (SEQ - 1));
  float freq = __expf(-(float)j * 0.28782313662425575f);  // ln(1e4)/32
  float sn, cs;
  __sincosf(t * freq, &sn, &cs);
  size_t base = (size_t)row * DMODEL + col;
  float x0 = in[base], x1 = in[base + 1];
  out[base]     = (bf16)(x0 * cs - x1 * sn);
  out[base + 1] = (bf16)(x1 * cs + x0 * sn);
}

// ---------------------------------------------------------------------------
// Lambda scalar: exp(dot(lq1,lk1)) - exp(dot(lq2,lk2)) + LAMBDA_INIT
// ---------------------------------------------------------------------------
__global__ void k_lambda(const float* __restrict__ lq1, const float* __restrict__ lk1,
                         const float* __restrict__ lq2, const float* __restrict__ lk2,
                         float* __restrict__ lam_out) {
  int lane = threadIdx.x;
  float s1 = lq1[lane] * lk1[lane] + lq1[lane + 32] * lk1[lane + 32];
  float s2 = lq2[lane] * lk2[lane] + lq2[lane + 32] * lk2[lane + 32];
  for (int off = 1; off < 32; off <<= 1) {
    s1 += __shfl_xor(s1, off, 32);
    s2 += __shfl_xor(s2, off, 32);
  }
  if (lane == 0) lam_out[0] = __expf(s1) - __expf(s2) + LAMBDA_INIT_F;
}

// ---------------------------------------------------------------------------
// Causal flash attention: 4-wave block covers 64 q-rows of one
// (batch, head, component). K tile (32x64) + V tile (128x32) staged in LDS by
// the Tensor Data Mover (wave 0 issues, TENSORcnt) with hardware row padding
// (K: 32dw+4dw pad -> 144B pitch; V: 16dw+4dw pad -> 80B pitch), double
// buffered, shared by all 4 waves. Falls back to per-lane async loads when
// the TDM builtin is unavailable.
// ---------------------------------------------------------------------------
__device__ __forceinline__ void flash_fill(const bf16* __restrict__ Kbase,
                                           const bf16* __restrict__ Vbase,
                                           int kb, char* Kbuf, char* Vbuf,
                                           int tid, int wave) {
#if HAVE_TDM
  if (wave == 0) {
    // K tile: 32 rows (k positions) x 64 dims; row stride DMODEL elements.
    tdm_load_2d((uint32_t)(uintptr_t)Kbuf, Kbase + (size_t)kb * DMODEL,
                /*tile_d0=*/HDQK, /*tile_d1=*/32,
                /*tensor_d0=*/HDQK, /*tensor_d1=*/SEQ,
                /*stride0=*/DMODEL,
                /*pad_interval(32dw)=*/4, /*pad_amount(4dw)=*/3);
    // V tile: 128 rows (v dims) x 32 k; row stride SEQ elements.
    tdm_load_2d((uint32_t)(uintptr_t)Vbuf, Vbase + kb,
                /*tile_d0=*/32, /*tile_d1=*/HDV,
                /*tensor_d0=*/32, /*tensor_d1=*/HDV,
                /*stride0=*/SEQ,
                /*pad_interval(16dw)=*/3, /*pad_amount(4dw)=*/3);
  }
#else
  const uint32_t lk = (uint32_t)(uintptr_t)Kbuf;
#pragma unroll
  for (int half = 0; half < 2; ++half) {
    const int ci = tid + half * 128;           // 256 chunks, 8 per K row
    const int row = ci >> 3, inrow = (ci & 7) * 16;
    const char* g = (const char*)(Kbase + (size_t)(kb + row) * DMODEL) + inrow;
    async_b128(lk + row * KPITCH + inrow, g);
  }
  const uint32_t lv = (uint32_t)(uintptr_t)Vbuf;
#pragma unroll
  for (int q = 0; q < 4; ++q) {
    const int ci = tid + q * 128;              // 512 chunks, 4 per V row
    const int row = ci >> 2, inrow = (ci & 3) * 16;
    const char* g = (const char*)(Vbase + (size_t)row * SEQ + kb) + inrow;
    async_b128(lv + row * VPITCH + inrow, g);
  }
#endif
}

__device__ __forceinline__ void flash_wait(int wave) {
#if HAVE_TDM
  if (wave == 0) wait_tensorcnt0();
#else
  wait_asynccnt0();
#endif
}

__global__ void __launch_bounds__(128)
k_flash(const bf16* __restrict__ Qr, const bf16* __restrict__ Kr,
        const bf16* __restrict__ Vt,
        float* __restrict__ O1, float* __restrict__ O2) {
  __shared__ char KbufS[2 * KTILE_B];          //  9 KB
  __shared__ char VbufS[2 * VTILE_B];          // 20 KB
  __shared__ float Pls[4][16 * 32];            //  8 KB (per-wave P regions)
  const int tid  = threadIdx.x;
  const int lane = tid & 31;
  const int wave = tid >> 5;
  const int q0b  = blockIdx.x * 64;
  const int q0   = q0b + wave * 16;
  const int comp = blockIdx.y & 1;
  const int h    = blockIdx.y >> 1;
  const int b    = blockIdx.z;
  float* O = comp ? O2 : O1;
  float* Pw = &Pls[wave][0];

  const bf16* Qbase = Qr + (size_t)b * SEQ * DMODEL + (size_t)(2 * h + comp) * HDQK;
  const bf16* Kbase = Kr + (size_t)b * SEQ * DMODEL + (size_t)(2 * h + comp) * HDQK;
  const bf16* Vbase = Vt + ((size_t)b * NHEADS + h) * HDV * SEQ;

  // Preload Q fragments (hd=64 -> two K=32 chunks)
  v16bf qfrag[2];
  {
    const int row = q0 + (lane & 15);
    const bf16* qp = Qbase + (size_t)row * DMODEL;
    const int kb = (lane >> 4) << 3;
#pragma unroll
    for (int c = 0; c < 2; ++c) {
      v8bf lo = *(const v8bf*)(qp + 32 * c + kb);
      v8bf hi = *(const v8bf*)(qp + 32 * c + kb + 16);
      qfrag[c] = cat8(lo, hi);
    }
  }

  v8f acc[8] = {};
  float mrow[8], lrow[8];
#pragma unroll
  for (int r = 0; r < 8; ++r) { mrow[r] = -INFINITY; lrow[r] = 0.0f; }

  const int rowoff = q0 + ((lane >> 4) << 3);
  const int kmax = q0b + 64;                   // block-wide causal limit

  flash_fill(Kbase, Vbase, 0, KbufS, VbufS, tid, wave);
  int c = 0;
  for (int kb = 0; kb < kmax; kb += 32) {
    flash_wait(wave);
    __syncthreads();                           // tiles (buffer c) ready
    if (kb + 32 < kmax)
      flash_fill(Kbase, Vbase, kb + 32,
                 KbufS + (1 - c) * KTILE_B, VbufS + (1 - c) * VTILE_B,
                 tid, wave);
    const char* Ktile = KbufS + c * KTILE_B;
    const char* Vtile = VbufS + c * VTILE_B;

    // ---- S = Q K^T for 16 q-rows x 32 k-cols (two 16x16 C tiles) ----
    v8f S[2] = {};
#pragma unroll
    for (int j = 0; j < 2; ++j)
#pragma unroll
      for (int c2 = 0; c2 < 2; ++c2)
        S[j] = wmma_bf16(qfrag[c2],
                         lds_b_frag(Ktile, 16 * j, c2 * 64, KPITCH, lane), S[j]);

    // ---- scale + causal mask ----
#pragma unroll
    for (int j = 0; j < 2; ++j) {
      const int col = kb + 16 * j + (lane & 15);
#pragma unroll
      for (int r = 0; r < 8; ++r) {
        float s = S[j][r] * 0.125f;            // 64^-0.5
        S[j][r] = (col <= rowoff + r) ? s : -INFINITY;
      }
    }
    // ---- online softmax (row reductions across 16-lane halves) ----
    float alpha[8];
#pragma unroll
    for (int r = 0; r < 8; ++r) {
      float t = fmaxf(S[0][r], S[1][r]);
      for (int off = 1; off < 16; off <<= 1) t = fmaxf(t, __shfl_xor(t, off, 32));
      const float mn = fmaxf(mrow[r], t);
      alpha[r] = __expf(mrow[r] - mn);
      mrow[r] = mn;
      const float p0 = __expf(S[0][r] - mn);
      const float p1 = __expf(S[1][r] - mn);
      S[0][r] = p0; S[1][r] = p1;
      float rs = p0 + p1;
      for (int off = 1; off < 16; off <<= 1) rs += __shfl_xor(rs, off, 32);
      lrow[r] = lrow[r] * alpha[r] + rs;
    }
#pragma unroll
    for (int j = 0; j < 8; ++j)
#pragma unroll
      for (int r = 0; r < 8; ++r) acc[j][r] *= alpha[r];

    // ---- stage P through per-wave LDS region (DS ops are in-order within a
    //      wave; no block barrier needed) ----
    {
      const int rb = (lane >> 4) << 3;
#pragma unroll
      for (int r = 0; r < 8; ++r) {
        Pw[(rb + r) * 32 + (lane & 15)]      = S[0][r];
        Pw[(rb + r) * 32 + 16 + (lane & 15)] = S[1][r];
      }
    }
    v16bf pfrag;
    {
      const int row = lane & 15;
      const int k2  = (lane >> 4) << 3;
      const float* pp = &Pw[row * 32 + k2];
#pragma unroll
      for (int i = 0; i < 8; ++i) {
        pfrag[i]     = (bf16)pp[i];
        pfrag[i + 8] = (bf16)pp[i + 16];
      }
    }
    // ---- O += P V  (dv=128 -> 8 N tiles) ----
#pragma unroll
    for (int j = 0; j < 8; ++j)
      acc[j] = wmma_bf16(pfrag,
                         lds_b_frag(Vtile, 16 * j, 0, VPITCH, lane), acc[j]);

    __syncthreads();                           // all reads of buffer c done
    c ^= 1;
  }

  // ---- epilogue: normalize and store ----
  const int rbase = q0 + ((lane >> 4) << 3);
#pragma unroll
  for (int j = 0; j < 8; ++j) {
    const int col = 16 * j + (lane & 15);
    float* op = O + ((size_t)b * SEQ + rbase) * DMODEL + h * HDV + col;
#pragma unroll
    for (int r = 0; r < 8; ++r) op[(size_t)r * DMODEL] = acc[j][r] / lrow[r];
  }
}

// ---------------------------------------------------------------------------
// diff + RMS-norm over dv=128, scale, -> bf16 ; one wave per (b,s,h) group
// ---------------------------------------------------------------------------
__global__ void k_combine(const float* __restrict__ A1, const float* __restrict__ A2,
                          const float* __restrict__ lam_p,
                          const float* __restrict__ ln_w, const float* __restrict__ ln_b,
                          bf16* __restrict__ outC) {
  const int lane = threadIdx.x;
  const int g = blockIdx.x;                 // (b*2048+s)*16 + h
  const int row = g >> 4, h = g & 15;
  const float lam = lam_p[0];
  const size_t base = (size_t)row * DMODEL + h * HDV;
  float d[4], ss = 0.0f;
#pragma unroll
  for (int i = 0; i < 4; ++i) {
    const int e = i * 32 + lane;
    const float v = A1[base + e] - lam * A2[base + e];
    d[i] = v; ss += v * v;
  }
  for (int off = 1; off < 32; off <<= 1) ss += __shfl_xor(ss, off, 32);
  const float rn = rsqrtf(ss * (1.0f / 128.0f) + 1e-8f);
#pragma unroll
  for (int i = 0; i < 4; ++i) {
    const int e = i * 32 + lane;
    outC[base + e] = (bf16)((d[i] * rn * ln_w[e] + ln_b[e]) * ONE_MINUS_LI_F);
  }
}

// ---------------------------------------------------------------------------
// Host launcher
// ---------------------------------------------------------------------------
#define MB(x) ((size_t)(x) * 1024u * 1024u)

extern "C" void kernel_launch(void* const* d_in, const int* in_sizes, int n_in,
                              void* d_out, int out_size, void* d_ws, size_t ws_size,
                              hipStream_t stream) {
  const float* x   = (const float*)d_in[0];
  const float* Wq  = (const float*)d_in[1];
  const float* Wk  = (const float*)d_in[2];
  const float* Wv  = (const float*)d_in[3];
  const float* Wo  = (const float*)d_in[4];
  const float* lq1 = (const float*)d_in[5];
  const float* lk1 = (const float*)d_in[6];
  const float* lq2 = (const float*)d_in[7];
  const float* lk2 = (const float*)d_in[8];
  const float* lnw = (const float*)d_in[9];
  const float* lnb = (const float*)d_in[10];
  float* out = (float*)d_out;

  char* ws = (char*)d_ws;
  bf16*  xb    = (bf16*)(ws);              // 16 MB
  bf16*  Wqt   = (bf16*)(ws + MB(16));     //  8 MB each
  bf16*  Wkt   = (bf16*)(ws + MB(24));
  bf16*  Wvt   = (bf16*)(ws + MB(32));
  bf16*  Wot   = (bf16*)(ws + MB(40));
  float* Qf    = (float*)(ws + MB(48));    // 32 MB (reused as attn1)
  float* Kf    = (float*)(ws + MB(80));    // 32 MB (reused as attn2)
  bf16*  Qr    = (bf16*)(ws + MB(112));    // 16 MB
  bf16*  Kr    = (bf16*)(ws + MB(128));    // 16 MB
  bf16*  Vt    = (bf16*)(ws + MB(144));    // 16 MB
  bf16*  attnC = (bf16*)(ws + MB(160));    // 16 MB
  float* lam   = (float*)(ws + MB(176));   //  4 B
  float* attn1 = Qf;
  float* attn2 = Kf;

  const int NTOT = NROWS * DMODEL;                       // 8M
  k_f32_to_bf16<<<NTOT / 256, 256, 0, stream>>>(x, xb, NTOT);
  dim3 tgrid(DMODEL / 32, DMODEL / 32), tblk(32, 8);
  k_transpose_w<<<tgrid, tblk, 0, stream>>>(Wq, Wqt);
  k_transpose_w<<<tgrid, tblk, 0, stream>>>(Wk, Wkt);
  k_transpose_w<<<tgrid, tblk, 0, stream>>>(Wv, Wvt);
  k_transpose_w<<<tgrid, tblk, 0, stream>>>(Wo, Wot);

  dim3 ggrid(NROWS / 64, DMODEL / 64);
  k_gemm_f32<<<ggrid, 128, 0, stream>>>(xb, Wqt, Qf);
  k_gemm_f32<<<ggrid, 128, 0, stream>>>(xb, Wkt, Kf);
  k_gemm_vt <<<ggrid, 128, 0, stream>>>(xb, Wvt, Vt);

  k_rope<<<(NROWS * 1024) / 256, 256, 0, stream>>>(Qf, Qr);
  k_rope<<<(NROWS * 1024) / 256, 256, 0, stream>>>(Kf, Kr);

  k_lambda<<<1, 32, 0, stream>>>(lq1, lk1, lq2, lk2, lam);

  dim3 fgrid(SEQ / 64, NHEADS * 2, BATCH);
  k_flash<<<fgrid, 128, 0, stream>>>(Qr, Kr, Vt, attn1, attn2);

  k_combine<<<NROWS * NHEADS, 32, 0, stream>>>(attn1, attn2, lam, lnw, lnb, attnC);

  k_gemm_f32<<<ggrid, 128, 0, stream>>>(attnC, Wot, out);
}
